// RBLNBGEM3ComputeScoreWrapper_48137993453786
// MI455X (gfx1250) — matrix operand order repaired and checked by reading, not compiled, not run on hardware
//
#include <hip/hip_runtime.h>
#include <hip/hip_bf16.h>

#define B_   16
#define S_   2048
#define H_   1024
#define V_   250002
#define EPSF 1e-12f

typedef __attribute__((ext_vector_type(16))) __bf16 v16bf;
typedef __attribute__((ext_vector_type(8)))  float  v8f;

union ABFrag { v16bf v; unsigned int u[8]; };

// Pack two f32 into two bf16 (round-half-up) with a single v_perm_b32:
// selector 0x07060302 picks bytes {S0[3],S0[2],S1[3],S1[2]} = {hi>>16, lo>>16}.
__device__ __forceinline__ unsigned int pack2_bf16(float lo, float hi) {
    const unsigned int a = __float_as_uint(lo) + 0x8000u;
    const unsigned int b = __float_as_uint(hi) + 0x8000u;
    return __builtin_amdgcn_perm(b, a, 0x07060302u);
}

// ---------------------------------------------------------------------------
// dense: L2-normalized CLS token per batch.  One wave per batch row.
// ---------------------------------------------------------------------------
__global__ __launch_bounds__(32) void dense_kernel(const float* __restrict__ hid,
                                                   float* __restrict__ out) {
    const int b    = blockIdx.x;
    const int lane = threadIdx.x;
    const float* p = hid + (size_t)b * S_ * H_;   // token s=0
    float vals[32];
    float ss = 0.f;
    #pragma unroll
    for (int i = 0; i < 32; ++i) { vals[i] = p[lane + i * 32]; ss += vals[i] * vals[i]; }
    #pragma unroll
    for (int off = 16; off; off >>= 1) ss += __shfl_xor(ss, off, 32);
    const float inv = 1.f / fmaxf(sqrtf(ss), EPSF);
    #pragma unroll
    for (int i = 0; i < 32; ++i) out[(size_t)b * H_ + lane + i * 32] = vals[i] * inv;
}

// ---------------------------------------------------------------------------
// sparse: zero buffer, per-token relu(matvec) scatter-max, clear special ids
// ---------------------------------------------------------------------------
__global__ __launch_bounds__(256) void zero_kernel(float* __restrict__ p, int n) {
    int i = blockIdx.x * 256 + threadIdx.x;
    if (i < n) p[i] = 0.f;
}

__global__ __launch_bounds__(256) void sparse_kernel(const float* __restrict__ hid,
                                                     const float* __restrict__ sW,
                                                     const float* __restrict__ sb,
                                                     const int*   __restrict__ ids,
                                                     float*       __restrict__ sp) {
    const int m    = blockIdx.x * 8 + (threadIdx.x >> 5);   // token index (b*S + s)
    const int lane = threadIdx.x & 31;
    if (m >= B_ * S_) return;
    const float* p = hid + (size_t)m * H_;
    float s = 0.f;
    #pragma unroll
    for (int i = 0; i < 32; ++i) s += p[lane + i * 32] * sW[lane + i * 32];
    #pragma unroll
    for (int off = 16; off; off >>= 1) s += __shfl_xor(s, off, 32);
    if (lane == 0) {
        const float w  = fmaxf(s + sb[0], 0.f);             // relu -> non-negative
        const int   b  = m >> 11;                            // / S_
        const int   id = ids[m];
        // non-negative float max == unsigned-int max on the bit pattern
        atomicMax((unsigned int*)&sp[(size_t)b * V_ + id], __float_as_uint(w));
    }
}

__global__ void sparse_clear_special(float* __restrict__ sp) {
    const int b = threadIdx.x >> 2;   // 16 batches x 4 special ids {0,1,2,3}
    const int t = threadIdx.x & 3;
    sp[(size_t)b * V_ + t] = 0.f;
}

// ---------------------------------------------------------------------------
// colbert GEMM: (B*S, H) x (H, H), bf16 WMMA, fp32 accumulate.
// Workgroup tile 128(M) x 128(N), K-step 32; 8 waves, wave tile 32x64
// (2x4 sub-tiles -> 8 v_wmma per K-step).  Register double-buffered staging:
// global loads for tile kt+32 are issued before the WMMAs for tile kt.
// Epilogue: +bias, *mask, store shifted to (b, s-1, :).
// ---------------------------------------------------------------------------
#define TM 128
#define TN 128
#define TK 32
#define LDA 36   // padded LDS row stride (ushorts): 72B rows -> 8B-aligned b64
#define LDB 36   // stores; 18-dword stride -> conflict-free fragment reads

__global__ __launch_bounds__(256) void colbert_gemm(const float* __restrict__ hid,
                                                    const float* __restrict__ W,     // [k][n]
                                                    const float* __restrict__ bias,
                                                    const float* __restrict__ mask,
                                                    float*       __restrict__ out) {
    __shared__ unsigned short Atile[TM * LDA];
    __shared__ unsigned short Btile[TN * LDB];   // transposed: Bt[n][k]

    const int tid   = threadIdx.x;
    const int lane  = tid & 31;
    const int wave  = tid >> 5;
    const int waveM = wave >> 1;                 // 0..3  (rows waveM*32..+31)
    const int waveN = wave & 1;                  // 0..1  (cols waveN*64..+63)
    const int half  = lane >> 4;                 // K-half select
    const int l15   = lane & 15;

    const int wgM = blockIdx.x * TM;             // base row (m = b*S + s)
    const int wgN = blockIdx.y * TN;             // base col

    v8f acc[2][4] = {};

    // register staging buffers (double-buffer vs. LDS)
    float4 ar[4];          // A: 4 tasks x float4 (row r, k-quad kq)
    float  br[4][4];       // B: 4 tasks x 4 k-consecutive floats (col n)

    auto loadA = [&](int kt) {
        #pragma unroll
        for (int p = 0; p < 4; ++p) {
            const int t = tid + p * 256;
            const int r = t >> 3, kq = t & 7;
            ar[p] = *(const float4*)&hid[(size_t)(wgM + r) * H_ + kt + kq * 4];
        }
    };
    auto loadB = [&](int kt) {
        #pragma unroll
        for (int p = 0; p < 4; ++p) {
            const int t = tid + p * 256;
            const int n = t & 127, kq = t >> 7;
            #pragma unroll
            for (int c = 0; c < 4; ++c)   // coalesced across lanes (consecutive n)
                br[p][c] = W[(size_t)(kt + kq * 4 + c) * H_ + wgN + n];
        }
    };
    auto storeTiles = [&]() {
        #pragma unroll
        for (int p = 0; p < 4; ++p) {
            const int t = tid + p * 256;
            const int r = t >> 3, kq = t & 7;
            uint2 d;
            d.x = pack2_bf16(ar[p].x, ar[p].y);
            d.y = pack2_bf16(ar[p].z, ar[p].w);
            *(uint2*)&Atile[r * LDA + kq * 4] = d;
        }
        #pragma unroll
        for (int p = 0; p < 4; ++p) {
            const int t = tid + p * 256;
            const int n = t & 127, kq = t >> 7;
            uint2 d;
            d.x = pack2_bf16(br[p][0], br[p][1]);
            d.y = pack2_bf16(br[p][2], br[p][3]);
            *(uint2*)&Btile[n * LDB + kq * 4] = d;
        }
    };

    loadA(0);
    loadB(0);

    for (int kt = 0; kt < H_; kt += TK) {
        storeTiles();
        __syncthreads();

        if (kt + TK < H_) {                       // overlap DRAM with matrix pipe
            loadA(kt + TK);
            loadB(kt + TK);
            __builtin_prefetch(&W[(size_t)(kt + TK + lane) * H_ + wgN], 0, 0);
        }

        // A fragments (ISA 16-bit A layout): lane l15 = row; half selects
        // K dword groups {0..3 | 4..7} and {8..11 | 12..15} of the 16-dword row
        ABFrag a[2], bfr[4];
        #pragma unroll
        for (int i = 0; i < 2; ++i) {
            const unsigned int* Arow =
                (const unsigned int*)&Atile[(waveM * 32 + i * 16 + l15) * LDA];
            #pragma unroll
            for (int j = 0; j < 4; ++j) {
                a[i].u[j]     = Arow[half * 4 + j];
                a[i].u[4 + j] = Arow[8 + half * 4 + j];
            }
        }
        // B fragments: lane l15 = column; half selects K dword groups {0..3 | 4..7} and {8..11 | 12..15} (same as A)
        #pragma unroll
        for (int j = 0; j < 4; ++j) {
            const unsigned int* Brow =
                (const unsigned int*)&Btile[(waveN * 64 + j * 16 + l15) * LDB];
            #pragma unroll
            for (int q = 0; q < 8; ++q) bfr[j].u[q] = Brow[(q >> 2) * 8 + half * 4 + (q & 3)];
        }

        #pragma unroll
        for (int i = 0; i < 2; ++i)
            #pragma unroll
            for (int j = 0; j < 4; ++j)
                acc[i][j] = __builtin_amdgcn_wmma_f32_16x16x32_bf16(
                    false, a[i].v, false, bfr[j].v, (short)0, acc[i][j], false, false);

        __syncthreads();
    }

    // epilogue: +bias, *mask, shifted store (skip s==0 rows)
    #pragma unroll
    for (int i = 0; i < 2; ++i) {
        #pragma unroll
        for (int j = 0; j < 4; ++j) {
            const int   n  = wgN + waveN * 64 + j * 16 + l15;
            const float bn = bias[n];
            #pragma unroll
            for (int v = 0; v < 8; ++v) {
                const int m = wgM + waveM * 32 + i * 16 + half * 8 + v;
                const int b = m >> 11;           // / S_
                const int s = m & (S_ - 1);
                if (s == 0) continue;
                const float mv = mask[b * S_ + s];
                out[((size_t)b * (S_ - 1) + (s - 1)) * H_ + n] = (acc[i][j][v] + bn) * mv;
            }
        }
    }
}

// Second pass: in-place row L2 normalization over the full H=1024 columns.
__global__ __launch_bounds__(256) void colbert_norm(float* __restrict__ out) {
    const int row  = blockIdx.x * 8 + (threadIdx.x >> 5);
    const int lane = threadIdx.x & 31;
    if (row >= B_ * (S_ - 1)) return;
    float* p = out + (size_t)row * H_;
    float vals[32];
    float ss = 0.f;
    #pragma unroll
    for (int i = 0; i < 32; ++i) { vals[i] = p[lane + i * 32]; ss += vals[i] * vals[i]; }
    #pragma unroll
    for (int off = 16; off; off >>= 1) ss += __shfl_xor(ss, off, 32);
    const float inv = 1.f / fmaxf(sqrtf(ss), EPSF);
    #pragma unroll
    for (int i = 0; i < 32; ++i) p[lane + i * 32] = vals[i] * inv;
}

// ---------------------------------------------------------------------------
extern "C" void kernel_launch(void* const* d_in, const int* in_sizes, int n_in,
                              void* d_out, int out_size, void* d_ws, size_t ws_size,
                              hipStream_t stream) {
    (void)in_sizes; (void)n_in; (void)out_size; (void)d_ws; (void)ws_size;

    const float* hid  = (const float*)d_in[0];   // (16, 2048, 1024) f32
    const float* mask = (const float*)d_in[1];   // (16, 2048) f32
    const int*   ids  = (const int*)d_in[2];     // (16, 2048) int
    const float* sW   = (const float*)d_in[3];   // (1024, 1) f32
    const float* sb   = (const float*)d_in[4];   // (1,) f32
    const float* cW   = (const float*)d_in[5];   // (1024, 1024) f32
    const float* cb   = (const float*)d_in[6];   // (1024,) f32

    float* out    = (float*)d_out;
    float* dense  = out;                                   // 16*1024
    float* sparse = dense + (size_t)B_ * H_;               // 16*250002
    float* colb   = sparse + (size_t)B_ * V_;              // 16*2047*1024

    dense_kernel<<<B_, 32, 0, stream>>>(hid, dense);

    const int nsp = B_ * V_;
    zero_kernel<<<(nsp + 255) / 256, 256, 0, stream>>>(sparse, nsp);
    sparse_kernel<<<(B_ * S_) / 8, 256, 0, stream>>>(hid, sW, sb, ids, sparse);
    sparse_clear_special<<<1, 64, 0, stream>>>(sparse);

    dim3 ggrid((B_ * S_) / TM, H_ / TN);                   // 256 x 8
    colbert_gemm<<<ggrid, 256, 0, stream>>>(hid, cW, cb, mask, colb);

    const int nrows = B_ * (S_ - 1);
    colbert_norm<<<(nrows + 7) / 8, 256, 0, stream>>>(colb);
}
